// LinearAttention_41180146434124
// MI455X (gfx1250) — matrix operand order repaired and checked by reading, not compiled, not run on hardware
//
#include <hip/hip_runtime.h>
#include <hip/hip_bf16.h>
#include <cstdint>

// ---------------- CDNA5 WMMA types ----------------
typedef __attribute__((ext_vector_type(16))) _Float16 v16h;
typedef __attribute__((ext_vector_type(8)))  float    v8f;

#define HEADS 8
#define B_    4
#define N_    8192
#define C_    512
#define D_    64
#define ROWS  (B_*N_)          // 32768
#define NCHUNK 8               // K-split for kv-state stage
#define EPS_LN   1e-5f
#define EPS_NORM 1e-8f

// Workspace layout (bytes); all offsets comfortably aligned.
static constexpr size_t SZ_QKVW  = (size_t)3*C_*C_*2;          // f16 qkv weights
static constexpr size_t SZ_OUTW  = (size_t)C_*C_*2;            // f16 out weights
static constexpr size_t SZ_BIG   = (size_t)ROWS*C_*2;          // 32 MB activation plane
static constexpr size_t SZ_KVT   = (size_t)32*D_*D_*2;         // kv^T f16 per (b,h)
static constexpr size_t SZ_KVP   = (size_t)NCHUNK*32*D_*D_*4;  // fp32 kv partials
static constexpr size_t SZ_KSP   = (size_t)NCHUNK*32*D_*4;     // fp32 ksum partials
static constexpr size_t OFF_QKVW = 0;
static constexpr size_t OFF_OUTW = OFF_QKVW + SZ_QKVW;
static constexpr size_t OFF_XN   = OFF_OUTW + SZ_OUTW;         // xn, reused as attn
static constexpr size_t OFF_Q    = OFF_XN + SZ_BIG;
static constexpr size_t OFF_KT   = OFF_Q  + SZ_BIG;
static constexpr size_t OFF_VT   = OFF_KT + SZ_BIG;
static constexpr size_t OFF_KVT  = OFF_VT + SZ_BIG;
static constexpr size_t OFF_KVP  = OFF_KVT + SZ_KVT;
static constexpr size_t OFF_KSP  = OFF_KVP + SZ_KVP;
static constexpr size_t OFF_KSUM = OFF_KSP + SZ_KSP;           // 32*64 fp32

union FragU { v16h v; uint4 q[2]; };

// A-matrix 16x32 f16 fragment: lane&15 = M row; lane-half picks K chunks {0-7,16-23}/{8-15,24-31}
__device__ __forceinline__ v16h load_frag_a(const _Float16* row, int lane, int koff) {
  const _Float16* p = row + koff + ((lane & 16) ? 8 : 0);
  FragU f;
  f.q[0] = *reinterpret_cast<const uint4*>(p);
  f.q[1] = *reinterpret_cast<const uint4*>(p + 16);
  return f.v;
}
// B-matrix 32x16 f16 fragment: lane&15 = N col; lane-half picks K chunks {0-7,16-23}/{8-15,24-31}
__device__ __forceinline__ v16h load_frag_b(const _Float16* col, int lane, int koff) {
  const _Float16* p = col + koff + ((lane & 16) ? 8 : 0);
  FragU f;
  f.q[0] = *reinterpret_cast<const uint4*>(p);
  f.q[1] = *reinterpret_cast<const uint4*>(p + 16);
  return f.v;
}
__device__ __forceinline__ v8f wmma_f16(v16h a, v16h b, v8f c) {
  return __builtin_amdgcn_wmma_f32_16x16x32_f16(false, a, false, b, (short)0, c, false, false);
}

// ---------------- fp32 -> f16 weight conversion ----------------
__global__ void cvt_kernel(const float* __restrict__ in, _Float16* __restrict__ out, int n) {
  int i = blockIdx.x * blockDim.x + threadIdx.x;
  if (i < n) out[i] = (_Float16)in[i];
}

// ---------------- LayerNorm: one wave per row ----------------
__global__ void ln_kernel(const float* __restrict__ x, const float* __restrict__ g,
                          const float* __restrict__ be, _Float16* __restrict__ xn) {
  int wave = threadIdx.x >> 5, lane = threadIdx.x & 31;
  int row = blockIdx.x * 8 + wave;
  const float* xr = x + (size_t)row * C_;
  float vals[16];
  float s = 0.f, s2 = 0.f;
  #pragma unroll
  for (int j = 0; j < 16; ++j) { float v = xr[lane + j*32]; vals[j] = v; s += v; s2 += v*v; }
  #pragma unroll
  for (int o = 16; o > 0; o >>= 1) { s += __shfl_xor(s, o, 32); s2 += __shfl_xor(s2, o, 32); }
  float mu   = s  * (1.f / C_);
  float var  = s2 * (1.f / C_) - mu * mu;
  float rstd = rsqrtf(var + EPS_LN);
  _Float16* xo = xn + (size_t)row * C_;
  #pragma unroll
  for (int j = 0; j < 16; ++j) {
    int c = lane + j*32;
    xo[c] = (_Float16)((vals[j] - mu) * rstd * g[c] + be[c]);
  }
}

// ------- QKV GEMM, 2x2 register-blocked (wave owns 32x32 output), + elu+1 scatter -------
__global__ void qkv_kernel(const _Float16* __restrict__ xn, const _Float16* __restrict__ w,
                           const float* __restrict__ bias, _Float16* __restrict__ qh,
                           _Float16* __restrict__ kT, _Float16* __restrict__ vT) {
  int wave = threadIdx.x >> 5, lane = threadIdx.x & 31;
  int gw = blockIdx.x * 8 + wave;
  int tm = gw / 48, to = gw % 48;              // 1024 x 48 blocks of 32x32
  const _Float16* ar0 = xn + (size_t)(tm*32 + (lane & 15)) * C_;
  const _Float16* ar1 = ar0 + (size_t)16 * C_;
  const _Float16* bc0 = w  + (size_t)(to*32 + (lane & 15)) * C_;   // B[k=c,n=o] = W[o,c]
  const _Float16* bc1 = bc0 + (size_t)16 * C_;
  v8f acc00 = {}, acc01 = {}, acc10 = {}, acc11 = {};
  #pragma unroll
  for (int kk = 0; kk < C_/32; ++kk) {
    v16h a0 = load_frag_a(ar0, lane, kk*32);
    v16h a1 = load_frag_a(ar1, lane, kk*32);
    v16h b0 = load_frag_b(bc0, lane, kk*32);
    v16h b1 = load_frag_b(bc1, lane, kk*32);
    acc00 = wmma_f16(a0, b0, acc00);
    acc01 = wmma_f16(a0, b1, acc01);
    acc10 = wmma_f16(a1, b0, acc10);
    acc11 = wmma_f16(a1, b1, acc11);
  }
  int hi = (lane & 16) ? 8 : 0;
  #pragma unroll
  for (int oi = 0; oi < 2; ++oi) {
    int o = to*32 + oi*16 + (lane & 15);
    int seg = o >> 9, within = o & 511, head = within >> 6, di = within & 63;
    float bv = bias[o];
    #pragma unroll
    for (int mi = 0; mi < 2; ++mi) {
      const v8f& acc = (oi == 0) ? (mi == 0 ? acc00 : acc10)
                                 : (mi == 0 ? acc01 : acc11);
      #pragma unroll
      for (int r = 0; r < 8; ++r) {
        int nloc = tm*32 + mi*16 + r + hi;             // global row in [0, B*N)
        int bidx = nloc >> 13, n = nloc & (N_ - 1);
        int bh = bidx * HEADS + head;
        float val = acc[r] + bv;
        if (seg == 0) {                                 // q : elu+1, [bh][n][d]
          val = (val > 0.f) ? val + 1.f : __expf(val);
          qh[((size_t)bh*N_ + n)*D_ + di] = (_Float16)val;
        } else if (seg == 1) {                          // k : elu+1, [bh][d][n]
          val = (val > 0.f) ? val + 1.f : __expf(val);
          kT[((size_t)bh*D_ + di)*N_ + n] = (_Float16)val;
        } else {                                        // v : [bh][d][n]
          vT[((size_t)bh*D_ + di)*N_ + n] = (_Float16)val;
        }
      }
    }
  }
}

// ------- kv partials per (b,h,chunk): kv = K^T V, plus k_sum via ones-matrix WMMA -------
__global__ void kv_kernel(const _Float16* __restrict__ kT, const _Float16* __restrict__ vT,
                          float* __restrict__ kvp, float* __restrict__ ksp) {
  int wave = threadIdx.x >> 5, lane = threadIdx.x & 31;
  int bh = blockIdx.x >> 3, chunk = blockIdx.x & (NCHUNK - 1);
  int ti0 = wave >> 2, tj = wave & 3, ti1 = ti0 + 2;   // wave owns tiles (ti0,tj),(ti1,tj)
  size_t nbase = (size_t)chunk * (N_ / NCHUNK);
  const _Float16* a0 = kT + ((size_t)bh*D_ + ti0*16 + (lane & 15)) * N_ + nbase;
  const _Float16* a1 = kT + ((size_t)bh*D_ + ti1*16 + (lane & 15)) * N_ + nbase;
  const _Float16* bc = vT + ((size_t)bh*D_ + tj *16 + (lane & 15)) * N_ + nbase;
  bool do_sum = (tj == 0);                             // one tj-slice computes k_sum rows
  v16h ones;
  #pragma unroll
  for (int i = 0; i < 16; ++i) ones[i] = (_Float16)1.f;
  v8f acc0 = {}, acc1 = {}, s0 = {}, s1 = {};
  for (int kk = 0; kk < (N_/NCHUNK)/32; ++kk) {
    v16h af0 = load_frag_a(a0, lane, kk*32);
    v16h af1 = load_frag_a(a1, lane, kk*32);
    v16h bf  = load_frag_b(bc, lane, kk*32);
    acc0 = wmma_f16(af0, bf, acc0);
    acc1 = wmma_f16(af1, bf, acc1);
    if (do_sum) {                                      // row sums of K tile = k_sum partial
      s0 = wmma_f16(af0, ones, s0);
      s1 = wmma_f16(af1, ones, s1);
    }
  }
  int dj = tj*16 + (lane & 15);
  int hi = (lane & 16) ? 8 : 0;
  float* kb = kvp + ((size_t)(chunk*32 + bh))*4096 + dj;   // [chunk][bh][di][dj]
  #pragma unroll
  for (int r = 0; r < 8; ++r) {
    kb[(size_t)(ti0*16 + hi + r)*64] = acc0[r];
    kb[(size_t)(ti1*16 + hi + r)*64] = acc1[r];
  }
  if (do_sum && (lane & 15) == 0) {                    // lanes 0,16 hold column 0 of D
    float* sb = ksp + ((size_t)(chunk*32 + bh))*64;
    #pragma unroll
    for (int r = 0; r < 8; ++r) {
      sb[ti0*16 + hi + r] = s0[r];
      sb[ti1*16 + hi + r] = s1[r];
    }
  }
}

// ------- reduce chunk partials: kvT f16 (transposed) + ksum fp32 -------
__global__ void kvred_kernel(const float* __restrict__ kvp, const float* __restrict__ ksp,
                             _Float16* __restrict__ kvT, float* __restrict__ ksum) {
  int i = blockIdx.x * 256 + threadIdx.x;              // 131072 threads
  int bh = i >> 12, rem = i & 4095, dj = rem >> 6, di = rem & 63;
  float s = 0.f;
  #pragma unroll
  for (int c = 0; c < NCHUNK; ++c)
    s += kvp[((size_t)(c*32 + bh))*4096 + di*64 + dj];
  kvT[i] = (_Float16)s;                                // kvT[bh][dj][di]
  if (i < 32*64) {
    float t = 0.f;
    #pragma unroll
    for (int c = 0; c < NCHUNK; ++c) t += ksp[(size_t)c*32*64 + i];
    ksum[i] = t;
  }
}

// ---------------- out = (Q @ kv) / (q . k_sum + eps), write attn f16 ----------------
__global__ void attn_kernel(const _Float16* __restrict__ qh, const _Float16* __restrict__ kvT,
                            const float* __restrict__ ksum, _Float16* __restrict__ attn) {
  __shared__ float normbuf[8][16];
  int wave = threadIdx.x >> 5, lane = threadIdx.x & 31;
  int gw = blockIdx.x * 8 + wave;
  int bh = gw >> 9, tn = gw & 511;
  int bidx = bh >> 3, head = bh & 7;
  const _Float16* arow = qh + ((size_t)bh*N_ + tn*16 + (lane & 15)) * D_;
  int col = lane & 15;
  v16h a_lo = load_frag_a(arow, lane, 0);              // hoisted: shared by all 4 col-tiles
  v16h a_hi = load_frag_a(arow, lane, 32);
  v8f acc[4] = {{}, {}, {}, {}};
  #pragma unroll
  for (int tj = 0; tj < 4; ++tj) {
    const _Float16* bc = kvT + (size_t)bh*4096 + (size_t)(tj*16 + col)*64; // kv^T[dj][di]
    acc[tj] = wmma_f16(a_lo, load_frag_b(bc, lane, 0),  acc[tj]);
    acc[tj] = wmma_f16(a_hi, load_frag_b(bc, lane, 32), acc[tj]);
  }
  if (lane < 16) {
    int n = tn*16 + lane;
    const _Float16* qr = qh + ((size_t)bh*N_ + n) * D_;
    const float* ks = ksum + (size_t)bh*64;
    float s = 0.f;
    #pragma unroll
    for (int di = 0; di < 64; ++di) s += (float)qr[di] * ks[di];
    normbuf[wave][lane] = s + EPS_NORM;
  }
  __syncthreads();
  int hi = (lane & 16) ? 8 : 0;
  #pragma unroll
  for (int tj = 0; tj < 4; ++tj) {
    #pragma unroll
    for (int r = 0; r < 8; ++r) {
      int m = r + hi;
      int n = tn*16 + m;
      float o = acc[tj][r] / normbuf[wave][m];
      attn[((size_t)bidx*N_ + n)*C_ + head*D_ + tj*16 + col] = (_Float16)o;
    }
  }
}

// ------- final: attn @ out_w^T + out_b + residual, 2x2 register-blocked -------
__global__ void out_kernel(const _Float16* __restrict__ attn, const _Float16* __restrict__ w,
                           const float* __restrict__ bias, const float* __restrict__ x,
                           float* __restrict__ out) {
  int wave = threadIdx.x >> 5, lane = threadIdx.x & 31;
  int gw = blockIdx.x * 8 + wave;
  int tm = gw >> 4, tc = gw & 15;                      // 1024 x 16 blocks of 32x32
  const _Float16* ar0 = attn + (size_t)(tm*32 + (lane & 15)) * C_;
  const _Float16* ar1 = ar0 + (size_t)16 * C_;
  const _Float16* bc0 = w    + (size_t)(tc*32 + (lane & 15)) * C_;
  const _Float16* bc1 = bc0 + (size_t)16 * C_;
  v8f acc00 = {}, acc01 = {}, acc10 = {}, acc11 = {};
  #pragma unroll
  for (int kk = 0; kk < C_/32; ++kk) {
    v16h a0 = load_frag_a(ar0, lane, kk*32);
    v16h a1 = load_frag_a(ar1, lane, kk*32);
    v16h b0 = load_frag_b(bc0, lane, kk*32);
    v16h b1 = load_frag_b(bc1, lane, kk*32);
    acc00 = wmma_f16(a0, b0, acc00);
    acc01 = wmma_f16(a0, b1, acc01);
    acc10 = wmma_f16(a1, b0, acc10);
    acc11 = wmma_f16(a1, b1, acc11);
  }
  int hi = (lane & 16) ? 8 : 0;
  #pragma unroll
  for (int oi = 0; oi < 2; ++oi) {
    int co = tc*32 + oi*16 + (lane & 15);
    float bv = bias[co];
    #pragma unroll
    for (int mi = 0; mi < 2; ++mi) {
      const v8f& acc = (oi == 0) ? (mi == 0 ? acc00 : acc10)
                                 : (mi == 0 ? acc01 : acc11);
      #pragma unroll
      for (int r = 0; r < 8; ++r) {
        size_t row = (size_t)(tm*32 + mi*16 + r + hi);
        out[row*C_ + co] = acc[r] + bv + x[row*C_ + co];
      }
    }
  }
}

extern "C" void kernel_launch(void* const* d_in, const int* in_sizes, int n_in,
                              void* d_out, int out_size, void* d_ws, size_t ws_size,
                              hipStream_t stream) {
  (void)in_sizes; (void)n_in; (void)out_size; (void)ws_size;
  const float* x     = (const float*)d_in[0];
  const float* qkv_w = (const float*)d_in[1];
  const float* qkv_b = (const float*)d_in[2];
  const float* out_w = (const float*)d_in[3];
  const float* out_b = (const float*)d_in[4];
  const float* ln_g  = (const float*)d_in[5];
  const float* ln_b  = (const float*)d_in[6];
  float* out = (float*)d_out;
  char* ws = (char*)d_ws;

  _Float16* qkv_w_h = (_Float16*)(ws + OFF_QKVW);
  _Float16* out_w_h = (_Float16*)(ws + OFF_OUTW);
  _Float16* xn      = (_Float16*)(ws + OFF_XN);   // reused as attn after qkv GEMM
  _Float16* qh      = (_Float16*)(ws + OFF_Q);
  _Float16* kT      = (_Float16*)(ws + OFF_KT);
  _Float16* vT      = (_Float16*)(ws + OFF_VT);
  _Float16* kvT     = (_Float16*)(ws + OFF_KVT);
  float*    kvp     = (float*)   (ws + OFF_KVP);
  float*    ksp     = (float*)   (ws + OFF_KSP);
  float*    ksum    = (float*)   (ws + OFF_KSUM);

  cvt_kernel<<<(3*C_*C_ + 255)/256, 256, 0, stream>>>(qkv_w, qkv_w_h, 3*C_*C_);
  cvt_kernel<<<(C_*C_   + 255)/256, 256, 0, stream>>>(out_w, out_w_h, C_*C_);
  ln_kernel<<<ROWS/8, 256, 0, stream>>>(x, ln_g, ln_b, xn);
  // 1024 M-blocks x 48 O-blocks (32x32 per wave)
  qkv_kernel<<<(1024*48)/8, 256, 0, stream>>>(xn, qkv_w_h, qkv_b, qh, kT, vT);
  // 32 (b,h) x 8 K-chunks
  kv_kernel<<<32*NCHUNK, 256, 0, stream>>>(kT, vT, kvp, ksp);
  kvred_kernel<<<(32*D_*D_)/256, 256, 0, stream>>>(kvp, ksp, kvT, ksum);
  // 32 (b,h) x 512 N-tiles, 8 per block
  attn_kernel<<<(32*512)/8, 256, 0, stream>>>(qh, kvT, ksum, xn);
  // 1024 M-blocks x 16 C-blocks (32x32 per wave)
  out_kernel<<<(1024*16)/8, 256, 0, stream>>>(xn, out_w_h, out_b, x, out);
}
